// Transformer_52115133169754
// MI455X (gfx1250) — matrix operand order repaired and checked by reading, not compiled, not run on hardware
//
#include <hip/hip_runtime.h>
#include <hip/hip_bf16.h>

typedef _Float16 half_t;
typedef __attribute__((ext_vector_type(16))) _Float16 v16h;
typedef __attribute__((ext_vector_type(8)))  _Float16 v8h;
typedef __attribute__((ext_vector_type(4)))  _Float16 v4h;
typedef __attribute__((ext_vector_type(8)))  float    v8f;
typedef __attribute__((ext_vector_type(4)))  float    v4f;

#define DM   512
#define SEQ  1024
#define BSZ  2
#define NROW 2048          // BSZ*SEQ
#define LNEPS 1e-6f

// ---------------------------------------------------------------------------
// TDM availability (tensor_load_to_lds arity differs across toolchains)
// ---------------------------------------------------------------------------
#if defined(__has_builtin)
#if __has_builtin(__builtin_amdgcn_tensor_load_to_lds) && \
    __has_builtin(__builtin_amdgcn_s_wait_tensorcnt)
#define USE_TDM 1
#endif
#endif
#ifndef USE_TDM
#define USE_TDM 0
#endif

// ---------------------------------------------------------------------------
// WMMA helpers (gfx1250 wave32, v_wmma_f32_16x16x32_f16)
// ---------------------------------------------------------------------------
__device__ inline v8f wmma16(v16h a, v16h b, v8f c) {
  return __builtin_amdgcn_wmma_f32_16x16x32_f16(false, a, false, b, (short)0, c,
                                                false, false);
}

__device__ inline v16h cat16(v8h a, v8h b) {
  v16h r;
#pragma unroll
  for (int i = 0; i < 8; ++i) { r[i] = a[i]; r[8 + i] = b[i]; }
  return r;
}

// A fragment (16x32 f16): lane<16 holds K {0..7,16..23}, lane>=16 holds +8.
__device__ inline v16h frag_a(const half_t* rowp, int kbase) {
  const int hf = (threadIdx.x & 31) >> 4;
  v8h x0 = *(const v8h*)(rowp + kbase + hf * 8);
  v8h x1 = *(const v8h*)(rowp + kbase + 16 + hf * 8);
  return cat16(x0, x1);
}

// B fragment (32x16 f16): lane<16 holds K {0..7,16..23} of col N=lane, lane>=16 holds +8.
__device__ inline v16h frag_b(const half_t* rowp, int kbase) {
  const int hf = (threadIdx.x & 31) >> 4;
  v8h x0 = *(const v8h*)(rowp + kbase + hf * 8);
  v8h x1 = *(const v8h*)(rowp + kbase + 16 + hf * 8);
  return cat16(x0, x1);
}

// A fragment from f32 LDS data (probability matrix), converted to f16.
__device__ inline v16h frag_a_f32(const float* rowp, int kbase) {
  const int hf = (threadIdx.x & 31) >> 4;
  const float* p0 = rowp + kbase + hf * 8;
  const float* p1 = rowp + kbase + 16 + hf * 8;
  v4f a0 = *(const v4f*)p0;
  v4f a1 = *(const v4f*)(p0 + 4);
  v4f b0 = *(const v4f*)p1;
  v4f b1 = *(const v4f*)(p1 + 4);
  v16h r;
#pragma unroll
  for (int i = 0; i < 4; ++i) {
    r[i]      = (half_t)a0[i];
    r[4 + i]  = (half_t)a1[i];
    r[8 + i]  = (half_t)b0[i];
    r[12 + i] = (half_t)b1[i];
  }
  return r;
}

// ---------------------------------------------------------------------------
// Weight pre-pass: W[K,N] f32 -> Wt[N,K] f16 (transposed, WMMA-B friendly)
// ---------------------------------------------------------------------------
__global__ __launch_bounds__(256) void convert_w_kernel(const float* __restrict__ W,
                                                        half_t* __restrict__ Wt,
                                                        int K, int N) {
  int idx = blockIdx.x * 256 + threadIdx.x;   // grid covers K*N exactly
  int k = idx / N, n = idx - k * N;
  Wt[(size_t)n * K + k] = (half_t)W[idx];
}

// ---------------------------------------------------------------------------
// Tensor Data Mover: DMA one 128(N) x 32(K) f16 tile of Wt into LDS with
// 80-byte padded row stride (64B data + 16B pad).  D# per ISA ch.8.
// ---------------------------------------------------------------------------
#if USE_TDM
typedef unsigned int u32x4_t __attribute__((ext_vector_type(4)));
typedef int i32x4_t __attribute__((ext_vector_type(4)));
typedef int i32x8_t __attribute__((ext_vector_type(8)));

__device__ inline void tdm_load_tile(const half_t* gsrc, unsigned int lds_byte_off,
                                     int K, int N) {
  unsigned long long ga = (unsigned long long)(uintptr_t)gsrc;
  u32x4_t g0;
  g0[0] = 1u;                                   // count=1, user mode
  g0[1] = lds_byte_off;                         // lds_addr
  g0[2] = (unsigned int)ga;                     // global_addr[31:0]
  g0[3] = (unsigned int)((ga >> 32) & 0x01FFFFFFu) | 0x80000000u;  // addr[56:32] | type=2
  i32x8_t g1;
  // data_size=2B, pad_enable, pad_interval: 8<<3=64B, pad_amount: 4 dwords=16B
  g1[0] = (int)((1u << 16) | (1u << 20) | (3u << 22) | (3u << 25));
  g1[1] = (int)(((unsigned)K & 0xFFFFu) << 16);                       // dim0[15:0]
  g1[2] = (int)((((unsigned)K >> 16) & 0xFFFFu) |
                (((unsigned)N & 0xFFFFu) << 16));                     // dim0[31:16] | dim1[15:0]
  g1[3] = (int)((((unsigned)N >> 16) & 0xFFFFu) | (32u << 16));       // dim1[31:16] | tile_dim0=32
  g1[4] = 128;                                                        // tile_dim1=128
  g1[5] = K;                                                          // dim0_stride[31:0]
  g1[6] = 0;
  g1[7] = 0;
  i32x4_t z4 = {0, 0, 0, 0};
#if defined(__clang_major__) && (__clang_major__ >= 23)
  i32x8_t z8 = {0, 0, 0, 0, 0, 0, 0, 0};
  __builtin_amdgcn_tensor_load_to_lds(g0, g1, z4, z4, z8, 0);
#else
  __builtin_amdgcn_tensor_load_to_lds(g0, g1, z4, z4, 0);
#endif
}
#endif

// ---------------------------------------------------------------------------
// GEMM: C[M,N] = A[M,K] * Wt[N,K]^T  (+bias)(+relu). A f32, W f16, f32 out.
// 128x128x32 tiles, 256 threads (8 waves): wave grid 4(M) x 2(N), 32x64/wave.
// B tiles arrive via TDM (double-buffered) when available.
// ---------------------------------------------------------------------------
#define KP 40  // padded K stride (halfs) = 80 bytes, 16B aligned, bank-staggered

template <int EPI>  // 0=none, 1=+bias, 2=+bias+relu
__global__ __launch_bounds__(256) void gemm_kernel(const float* __restrict__ A,
                                                   const half_t* __restrict__ Bt,
                                                   const float* __restrict__ bias,
                                                   float* __restrict__ C,
                                                   int M, int N, int K) {
  __shared__ half_t As[128 * KP];
  __shared__ half_t Bs[2][128 * KP];

  const int tid  = threadIdx.x;
  const int lane = tid & 31;
  const int wv   = tid >> 5;
  const int r16  = lane & 15;
  const int hf   = lane >> 4;
  const int bm   = blockIdx.x * 128;
  const int bn   = blockIdx.y * 128;
  const int wm   = (wv & 3) * 32;   // wave M offset
  const int wn   = (wv >> 2) * 64;  // wave N offset

  const v8f vzero = {0.f, 0.f, 0.f, 0.f, 0.f, 0.f, 0.f, 0.f};
  v8f acc[2][4];
#pragma unroll
  for (int i = 0; i < 2; ++i)
#pragma unroll
    for (int j = 0; j < 4; ++j) acc[i][j] = vzero;

  const int ar = tid >> 3;        // A stage: row 0..31 (+32/pass)
  const int ac = (tid & 7) * 4;   // A stage: 4 consecutive K

  int cur = 0;
#if USE_TDM
  const unsigned int bs_off0 = (unsigned int)(uintptr_t)(void*)&Bs[0][0];
  const unsigned int bs_off1 = (unsigned int)(uintptr_t)(void*)&Bs[1][0];
  if (wv == 0) tdm_load_tile(Bt + (size_t)bn * K, bs_off0, K, N);
#endif

  for (int k0 = 0; k0 < K; k0 += 32) {
    __syncthreads();   // previous compute finished; LDS reusable
#if USE_TDM
    if (wv == 0 && k0 + 32 < K)   // prefetch next B tile into the other buffer
      tdm_load_tile(Bt + (size_t)bn * K + (k0 + 32), cur ? bs_off0 : bs_off1, K, N);
#endif
    // stage A tile (f32 -> f16)
#pragma unroll
    for (int it = 0; it < 4; ++it) {
      int r = ar + it * 32;
      v4f va = *(const v4f*)(A + (size_t)(bm + r) * K + k0 + ac);
      v4h ha = {(half_t)va[0], (half_t)va[1], (half_t)va[2], (half_t)va[3]};
      *(v4h*)(As + r * KP + ac) = ha;
    }
    if (k0 + 32 < K)
      __builtin_prefetch(A + (size_t)(bm + ar) * K + k0 + 32 + ac, 0, 1);

#if !USE_TDM
    // fallback: stage B tile with vector loads (Wt already transposed f16)
    {
      int n = tid >> 1, c = (tid & 1) * 16;
      const half_t* src = Bt + (size_t)(bn + n) * K + k0 + c;
      v8h w0 = *(const v8h*)src;
      v8h w1 = *(const v8h*)(src + 8);
      *(v8h*)(&Bs[0][0] + n * KP + c) = w0;
      *(v8h*)(&Bs[0][0] + n * KP + c + 8) = w1;
    }
#else
    if (wv == 0) {                  // current tile's DMA must have landed
      if (k0 + 32 < K) __builtin_amdgcn_s_wait_tensorcnt(1);
      else             __builtin_amdgcn_s_wait_tensorcnt(0);
    }
#endif
    __syncthreads();

    const half_t* Bsc = &Bs[cur][0];
    v16h af[2], bf[4];
#pragma unroll
    for (int i = 0; i < 2; ++i) af[i] = frag_a(As + (wm + i * 16 + r16) * KP, 0);
#pragma unroll
    for (int j = 0; j < 4; ++j) bf[j] = frag_b(Bsc + (wn + j * 16 + r16) * KP, 0);
#pragma unroll
    for (int i = 0; i < 2; ++i)
#pragma unroll
      for (int j = 0; j < 4; ++j) acc[i][j] = wmma16(af[i], bf[j], acc[i][j]);
#if USE_TDM
    cur ^= 1;
#endif
  }

#pragma unroll
  for (int j = 0; j < 4; ++j) {
    int col = bn + wn + j * 16 + r16;
    float bv = (EPI > 0) ? bias[col] : 0.f;
#pragma unroll
    for (int i = 0; i < 2; ++i) {
      int row0 = bm + wm + i * 16 + 8 * hf;
#pragma unroll
      for (int r = 0; r < 8; ++r) {
        float v = acc[i][j][r] + bv;
        if (EPI == 2) v = fmaxf(v, 0.f);
        C[(size_t)(row0 + r) * N + col] = v;
      }
    }
  }
}

// ---------------------------------------------------------------------------
// Fused attention per (batch, head, 64-query block). 256 threads (8 waves).
// Scores tile 64x1024 f32 resident in dynamic LDS (CDNA5 320KB WGP LDS).
// ---------------------------------------------------------------------------
#define AP 72  // padded stride (halfs) for q/k/v tiles = 144B, 16B aligned
#define ATT_SC   0
#define ATT_QS   (64 * SEQ * 4)            // 262144
#define ATT_KV   (ATT_QS + 64 * AP * 2)    // +9216
#define ATT_MB   (ATT_KV + 64 * AP * 2)    // +9216
#define ATT_RED  (ATT_MB + SEQ * 4)        // +4096
#define ATT_RED2 (ATT_RED + 256 * 4)
#define ATT_LDS  (ATT_RED2 + 256 * 4)      // = 286720 bytes

__global__ __launch_bounds__(256) void attn_kernel(const float* __restrict__ Q,
                                                   const float* __restrict__ Kx,
                                                   const float* __restrict__ Vx,
                                                   const int* __restrict__ mask,
                                                   float* __restrict__ O) {
  extern __shared__ char smem[];
  float*  sc   = (float*)(smem + ATT_SC);    // [64][1024]
  half_t* qs   = (half_t*)(smem + ATT_QS);   // [64][AP]
  half_t* kv   = (half_t*)(smem + ATT_KV);   // [64][AP] (K rows, then V^T rows)
  float*  mb   = (float*)(smem + ATT_MB);    // [1024] additive mask bias
  float*  red  = (float*)(smem + ATT_RED);   // [256]
  float*  red2 = (float*)(smem + ATT_RED2);  // [256]

  const int tid  = threadIdx.x;
  const int lane = tid & 31;
  const int wv   = tid >> 5;
  const int r16  = lane & 15;
  const int hf   = lane >> 4;
  const int blk  = blockIdx.x;
  const int qb   = blk & 15;
  const int h    = (blk >> 4) & 7;
  const int b    = blk >> 7;

  const float* Qb = Q  + ((size_t)b * SEQ + qb * 64) * DM + h * 64;
  const float* Kb = Kx + ((size_t)b * SEQ) * DM + h * 64;
  const float* Vb = Vx + ((size_t)b * SEQ) * DM + h * 64;

  for (int i = tid; i < SEQ; i += 256)
    mb[i] = (mask[b * SEQ + i] != 0) ? 0.f : -1e9f;

  // stage Q tile 64x64 f32 -> f16
#pragma unroll
  for (int it = 0; it < 4; ++it) {
    int r = (tid >> 4) + it * 16, c = (tid & 15) * 4;
    v4f x = *(const v4f*)(Qb + (size_t)r * DM + c);
    v4h hx = {(half_t)x[0], (half_t)x[1], (half_t)x[2], (half_t)x[3]};
    *(v4h*)(qs + r * AP + c) = hx;
  }

  // ---- S = (Q/8) K^T, masked, into sc ----
  const int wq = (wv & 3) * 16;   // wave q-row strip
  const int wk = (wv >> 2) * 32;  // wave key-col strip within tile
  for (int kt = 0; kt < 16; ++kt) {
    __syncthreads();
#pragma unroll
    for (int it = 0; it < 4; ++it) {  // stage K tile [key][d]
      int r = (tid >> 4) + it * 16, c = (tid & 15) * 4;
      v4f x = *(const v4f*)(Kb + (size_t)(kt * 64 + r) * DM + c);
      v4h hx = {(half_t)x[0], (half_t)x[1], (half_t)x[2], (half_t)x[3]};
      *(v4h*)(kv + r * AP + c) = hx;
    }
    __syncthreads();
    v16h a0 = frag_a(qs + (wq + r16) * AP, 0);
    v16h a1 = frag_a(qs + (wq + r16) * AP, 32);
#pragma unroll
    for (int n = 0; n < 2; ++n) {
      v16h b0 = frag_b(kv + (wk + n * 16 + r16) * AP, 0);
      v16h b1 = frag_b(kv + (wk + n * 16 + r16) * AP, 32);
      v8f cf = {0.f, 0.f, 0.f, 0.f, 0.f, 0.f, 0.f, 0.f};
      cf = wmma16(a0, b0, cf);
      cf = wmma16(a1, b1, cf);
      int colb = kt * 64 + wk + n * 16 + r16;
      float bias = mb[colb];
#pragma unroll
      for (int rr = 0; rr < 8; ++rr)
        sc[(size_t)(wq + rr + 8 * hf) * SEQ + colb] = cf[rr] * 0.125f + bias;
    }
  }
  __syncthreads();

  // ---- softmax over rows of sc (4 threads per row, 256-wide segments) ----
  {
    int row = tid >> 2, seg = tid & 3;
    float* rp = sc + (size_t)row * SEQ + seg * 256;
    float m = -3.0e38f;
    for (int i = 0; i < 256; ++i) m = fmaxf(m, rp[i]);
    red[tid] = m;
    __syncthreads();
    int rb = row * 4;
    m = fmaxf(fmaxf(red[rb], red[rb + 1]), fmaxf(red[rb + 2], red[rb + 3]));
    float s = 0.f;
    for (int i = 0; i < 256; ++i) { float e = __expf(rp[i] - m); rp[i] = e; s += e; }
    red2[tid] = s;
    __syncthreads();
    s = red2[rb] + red2[rb + 1] + red2[rb + 2] + red2[rb + 3];
    float inv = 1.f / s;
    for (int i = 0; i < 256; ++i) rp[i] *= inv;
  }

  // ---- O = P V ----
  const int wq2 = (wv & 3) * 16;
  const int wd  = (wv >> 2) * 32;
  v8f o0 = {0.f, 0.f, 0.f, 0.f, 0.f, 0.f, 0.f, 0.f};
  v8f o1 = o0;
  for (int kt = 0; kt < 16; ++kt) {
    __syncthreads();
#pragma unroll
    for (int it = 0; it < 4; ++it) {  // stage V tile transposed: [d][key]
      int key = (tid >> 4) + it * 16, c = (tid & 15) * 4;
      v4f x = *(const v4f*)(Vb + (size_t)(kt * 64 + key) * DM + c);
      kv[(c + 0) * AP + key] = (half_t)x[0];
      kv[(c + 1) * AP + key] = (half_t)x[1];
      kv[(c + 2) * AP + key] = (half_t)x[2];
      kv[(c + 3) * AP + key] = (half_t)x[3];
    }
    __syncthreads();
    const float* prow = sc + (size_t)(wq2 + r16) * SEQ + kt * 64;
#pragma unroll
    for (int ks = 0; ks < 2; ++ks) {
      v16h pa  = frag_a_f32(prow, ks * 32);
      v16h vb0 = frag_b(kv + (wd + r16) * AP, ks * 32);
      v16h vb1 = frag_b(kv + (wd + 16 + r16) * AP, ks * 32);
      o0 = wmma16(pa, vb0, o0);
      o1 = wmma16(pa, vb1, o1);
    }
  }
  float* Ob = O + ((size_t)b * SEQ + qb * 64) * DM + h * 64;
#pragma unroll
  for (int rr = 0; rr < 8; ++rr) {
    int r = wq2 + rr + 8 * hf;
    Ob[(size_t)r * DM + wd + r16]      = o0[rr];
    Ob[(size_t)r * DM + wd + 16 + r16] = o1[rr];
  }
}

// ---------------------------------------------------------------------------
// LayerNorm kernels (one 128-thread block per row of 512)
// ---------------------------------------------------------------------------
__device__ inline float blk_sum128(float x, float* red) {
#pragma unroll
  for (int o = 16; o > 0; o >>= 1) x += __shfl_down(x, o, 32);
  __syncthreads();
  if ((threadIdx.x & 31) == 0) red[threadIdx.x >> 5] = x;
  __syncthreads();
  return red[0] + red[1] + red[2] + red[3];
}

__global__ __launch_bounds__(128) void prep_ln_kernel(const float* __restrict__ x,
                                                      const float* __restrict__ pos,
                                                      const float* __restrict__ ti,
                                                      const float* __restrict__ g,
                                                      const float* __restrict__ bta,
                                                      float* __restrict__ out) {
  __shared__ float red[4];
  int row = blockIdx.x, tid = threadIdx.x;
  int l = row & (SEQ - 1);
  float t = ti ? ti[row] : 1.f;
  float v[4]; float s = 0.f;
#pragma unroll
  for (int i = 0; i < 4; ++i) {
    int c = tid + i * 128;
    v[i] = (x[(size_t)row * DM + c] + pos[(size_t)l * DM + c]) * t;
    s += v[i];
  }
  float mu = blk_sum128(s, red) * (1.f / DM);
  float q = 0.f;
#pragma unroll
  for (int i = 0; i < 4; ++i) { float d = v[i] - mu; q += d * d; }
  float var = blk_sum128(q, red) * (1.f / DM);
  float inv = rsqrtf(var + LNEPS);
#pragma unroll
  for (int i = 0; i < 4; ++i) {
    int c = tid + i * 128;
    out[(size_t)row * DM + c] = (v[i] - mu) * inv * g[c] + bta[c];
  }
}

__global__ __launch_bounds__(128) void resid_ln_kernel(const float* __restrict__ y,
                                                       const float* __restrict__ bias,
                                                       const float* __restrict__ resid,
                                                       const float* __restrict__ g,
                                                       const float* __restrict__ bta,
                                                       float* __restrict__ out) {
  __shared__ float red[4];
  int row = blockIdx.x, tid = threadIdx.x;
  float v[4]; float s = 0.f;
#pragma unroll
  for (int i = 0; i < 4; ++i) {
    int c = tid + i * 128;
    float xx = y[(size_t)row * DM + c] + resid[(size_t)row * DM + c];
    if (bias) xx += bias[c];
    v[i] = xx; s += xx;
  }
  float mu = blk_sum128(s, red) * (1.f / DM);
  float q = 0.f;
#pragma unroll
  for (int i = 0; i < 4; ++i) { float d = v[i] - mu; q += d * d; }
  float var = blk_sum128(q, red) * (1.f / DM);
  float inv = rsqrtf(var + LNEPS);
#pragma unroll
  for (int i = 0; i < 4; ++i) {
    int c = tid + i * 128;
    out[(size_t)row * DM + c] = (v[i] - mu) * inv * g[c] + bta[c];
  }
}

// ---------------------------------------------------------------------------
// Host orchestration
// ---------------------------------------------------------------------------
// Param flattening (jax pytree order: dict keys sorted):
//  d_in[0..4] = src_emb, src_mask_data, words_ti_value, trg_emb, trg_mask_data
//  then params: 'dec'(6 x ['enc','ffn','slf']), 'enc'(6 x ['ffn','slf']), 'ln'
//    mha dict sorted: fc, ln_b, ln_g, wk, wq, wv
//    ffn dict sorted: b1, b2, ln_b, ln_g, w1, w2
//    ln  dict sorted: dec_ln_b, dec_ln_g, enc_ln_b, enc_ln_g
//  pos_table last.
extern "C" void kernel_launch(void* const* d_in, const int* in_sizes, int n_in,
                              void* d_out, int out_size, void* d_ws, size_t ws_size,
                              hipStream_t stream) {
  (void)in_sizes; (void)n_in; (void)out_size; (void)ws_size;

  auto P = [&](int i) { return (const float*)d_in[i]; };
  const float* src   = P(0);
  const int*   smask = (const int*)d_in[1];
  const float* ti    = P(2);
  const float* trg   = P(3);
  const int*   tmask = (const int*)d_in[4];

  const int DEC0 = 5;                 // 6 * 18
  const int ENC0 = 5 + 108;           // 6 * 12
  const int LN0  = 5 + 108 + 72;      // 4
  const int POSI = LN0 + 4;
  const float* pos = P(POSI);

  const size_t NA = (size_t)NROW * DM;
  float* Xe = (float*)d_ws;
  float* Xd = Xe + NA;
  float* Qb = Xd + NA;
  float* Kb = Qb + NA;
  float* Vb = Kb + NA;
  float* Ab = Vb + NA;
  float* Tb = Ab + NA;
  float* Hb = Tb + NA;                       // 2048*2048 f32
  half_t* Wt = (half_t*)(Hb + (size_t)NROW * 2048);  // 2MB transposed f16 weight scratch

  auto gemm = [&](const float* A, const float* W, const float* bias, float* C,
                  int M, int N, int K, int epi) {
    convert_w_kernel<<<(K * N) / 256, 256, 0, stream>>>(W, Wt, K, N);
    dim3 g(M / 128, N / 128);
    if (epi == 2)      gemm_kernel<2><<<g, 256, 0, stream>>>(A, Wt, bias, C, M, N, K);
    else if (epi == 1) gemm_kernel<1><<<g, 256, 0, stream>>>(A, Wt, bias, C, M, N, K);
    else               gemm_kernel<0><<<g, 256, 0, stream>>>(A, Wt, nullptr, C, M, N, K);
  };

  // mp order: fc, ln_b, ln_g, wk, wq, wv
  auto run_mha = [&](int base, float* xq, const float* xkv, const int* mask, float* outX) {
    const float* fc  = P(base + 0);
    const float* lnb = P(base + 1);
    const float* lng = P(base + 2);
    const float* wk  = P(base + 3);
    const float* wq  = P(base + 4);
    const float* wvp = P(base + 5);
    gemm(xq,  wq,  nullptr, Qb, NROW, DM, DM, 0);
    gemm(xkv, wk,  nullptr, Kb, NROW, DM, DM, 0);
    gemm(xkv, wvp, nullptr, Vb, NROW, DM, DM, 0);
    attn_kernel<<<BSZ * 8 * (SEQ / 64), 256, ATT_LDS, stream>>>(Qb, Kb, Vb, mask, Ab);
    gemm(Ab, fc, nullptr, Tb, NROW, DM, DM, 0);
    resid_ln_kernel<<<NROW, 128, 0, stream>>>(Tb, nullptr, xq, lng, lnb, outX);
  };

  // fp order: b1, b2, ln_b, ln_g, w1, w2
  auto run_ffn = [&](int base, float* x, float* outX) {
    const float* b1  = P(base + 0);
    const float* b2  = P(base + 1);
    const float* lnb = P(base + 2);
    const float* lng = P(base + 3);
    const float* w1  = P(base + 4);
    const float* w2  = P(base + 5);
    gemm(x,  w1, b1,      Hb, NROW, 2048, DM,   2);
    gemm(Hb, w2, nullptr, Tb, NROW, DM,   2048, 0);
    resid_ln_kernel<<<NROW, 128, 0, stream>>>(Tb, b2, x, lng, lnb, outX);
  };

  // ---- encoder ----
  prep_ln_kernel<<<NROW, 128, 0, stream>>>(src, pos, ti, P(LN0 + 3), P(LN0 + 2), Xe);
  for (int l = 0; l < 6; ++l) {
    int eb = ENC0 + 12 * l;          // ffn @ +0, slf @ +6
    run_mha(eb + 6, Xe, Xe, smask, Xe);
    run_ffn(eb + 0, Xe, Xe);
  }

  // ---- decoder ----
  prep_ln_kernel<<<NROW, 128, 0, stream>>>(trg, pos, nullptr, P(LN0 + 1), P(LN0 + 0), Xd);
  for (int l = 0; l < 6; ++l) {
    int db = DEC0 + 18 * l;          // enc @ +0, ffn @ +6, slf @ +12
    run_mha(db + 12, Xd, Xd, tmask, Xd);   // self-attention
    run_mha(db + 0,  Xd, Xe, smask, Xd);   // cross-attention (K/V from encoder)
    run_ffn(db + 6,  Xd, (l == 5) ? (float*)d_out : Xd);
  }
}
